// Write_24979529793587
// MI455X (gfx1250) — matrix operand order repaired and checked by reading, not compiled, not run on hardware
//
#include <hip/hip_runtime.h>
#include <hip/hip_bf16.h>

// ---------------------------------------------------------------------------
// Pointer-network decoder for MI455X (gfx1250, wave32, WMMA).
// B=64, E=512 (unused), H=512, L=128.
// ---------------------------------------------------------------------------

typedef __bf16 bf16;
typedef __attribute__((ext_vector_type(16))) __bf16 v16bf;
typedef __attribute__((ext_vector_type(8)))  __bf16 v8bf;
typedef __attribute__((ext_vector_type(8)))  float  v8f;

#define LOG2E 1.4426950408889634f

constexpr int Bsz = 64;
constexpr int Hd  = 512;
constexpr int Ld  = 128;
constexpr int G4  = 4 * Hd;   // 2048
constexpr int K2H = 2 * Hd;   // 1024

// ---------------- fast math helpers (CDNA5 transcendental VALU) ------------
__device__ __forceinline__ float fexp2(float x) {
#if __has_builtin(__builtin_amdgcn_exp2f)
  return __builtin_amdgcn_exp2f(x);
#else
  return exp2f(x);
#endif
}
__device__ __forceinline__ float frcp(float x) {
#if __has_builtin(__builtin_amdgcn_rcpf)
  return __builtin_amdgcn_rcpf(x);
#else
  return 1.0f / x;
#endif
}
__device__ __forceinline__ float fsigm(float x) {
  // 1 / (1 + e^-x)
  return frcp(1.0f + fexp2(-x * LOG2E));
}
__device__ __forceinline__ float ftanh(float x) {
#if __has_builtin(__builtin_amdgcn_tanhf)
  return __builtin_amdgcn_tanhf(x);
#else
  // tanh(x) = 1 - 2/(e^{2x} + 1)
  return 1.0f - 2.0f * frcp(1.0f + fexp2(2.0f * LOG2E * x));
#endif
}

// ---------------- WMMA fragment loads (wave32 bf16 16x16x32) ---------------
// A fragment: lane holds row M = lane%16; VGPR0-3 = K[8h .. 8h+7],
// VGPR4-7 = K[16+8h .. 16+8h+7]  (h = lane/16). Two 16-byte chunks.
// Caller passes p = &A[row][k0 + 8*half]; chunks at p and p+16 elements.
__device__ __forceinline__ v16bf load_a_frag(const bf16* p) {
  v8bf lo = *(const v8bf*)p;
  v8bf hi = *(const v8bf*)(p + 16);
  v16bf r;
#pragma unroll
  for (int i = 0; i < 8; ++i) { r[i] = lo[i]; r[i + 8] = hi[i]; }
  return r;
}
// B fragment: lane holds column N = lane%16; K[8h..8h+7], K[16+8h..16+8h+7] (stride ld).
__device__ __forceinline__ v16bf load_b_frag(const bf16* p, int ld) {
  v16bf r;
#pragma unroll
  for (int i = 0; i < 8; ++i) { r[i] = p[i * ld]; r[i + 8] = p[(16 + i) * ld]; }
  return r;
}
#define WMMA_BF16(A, Bm, C) \
  __builtin_amdgcn_wmma_f32_16x16x32_bf16(false, (A), false, (Bm), (short)0, (C), false, false)

// ---------------------------------------------------------------------------
// Prep kernels
// ---------------------------------------------------------------------------
__global__ void k_transpose_bf(const float* __restrict__ src, bf16* __restrict__ dst,
                               int R, int C) {
  int i = blockIdx.x * blockDim.x + threadIdx.x;
  if (i < R * C) {
    int r = i / C, c = i % C;
    dst[c * R + r] = (bf16)src[i];   // dst is C x R
  }
}
__global__ void k_cvt_bf(const float* __restrict__ src, bf16* __restrict__ dst, int n) {
  int i = blockIdx.x * blockDim.x + threadIdx.x;
  if (i < n) dst[i] = (bf16)src[i];
}
__global__ void k_init_state(const float* __restrict__ h0, const float* __restrict__ c0,
                             float* __restrict__ h, float* __restrict__ c,
                             bf16* __restrict__ h_bf, float* __restrict__ mask) {
  int i = blockIdx.x * blockDim.x + threadIdx.x;
  if (i < Bsz * Hd) {
    float v = h0[i];
    h[i] = v;
    h_bf[i] = (bf16)v;
    c[i] = c0[i];
  }
  if (i < Bsz * Ld) mask[i] = 1.0f;
}
__global__ void k_finalize(const float* __restrict__ h, const float* __restrict__ c,
                           float* __restrict__ oh, float* __restrict__ oc) {
  int i = blockIdx.x * blockDim.x + threadIdx.x;
  if (i < Bsz * Hd) { oh[i] = h[i]; oc[i] = c[i]; }
}

// ---------------------------------------------------------------------------
// Precompute ctx[b,h,l] = sum_d Wctx[h,d]*context[b,d,l] + b_ctx[h]  (WMMA)
// grid: (L/16, H/16, B), block: 32 (one wave per 16x16 tile)
// ---------------------------------------------------------------------------
__global__ void __launch_bounds__(32)
k_ctx_gemm(const bf16* __restrict__ Wctx_bf, const bf16* __restrict__ ctxin_bf,
           const float* __restrict__ b_ctx, float* __restrict__ ctx) {
  int lane = threadIdx.x;
  int mrow = lane & 15, half = lane >> 4;
  int l0 = blockIdx.x * 16;
  int h0 = blockIdx.y * 16;
  int b  = blockIdx.z;
  const bf16* Bb = ctxin_bf + b * Hd * Ld;
  v8f acc = {};
  for (int k0 = 0; k0 < Hd; k0 += 32) {
    v16bf a  = load_a_frag(Wctx_bf + (h0 + mrow) * Hd + k0 + half * 8);
    v16bf bb = load_b_frag(Bb + (k0 + half * 8) * Ld + l0 + mrow, Ld);
    acc = WMMA_BF16(a, bb, acc);
  }
  float* outb = ctx + b * Hd * Ld;
  int col = l0 + mrow;
#pragma unroll
  for (int r = 0; r < 8; ++r) {
    int h = h0 + r + half * 8;
    outb[h * Ld + col] = acc[r] + b_ctx[h];
  }
}

// ---------------------------------------------------------------------------
// K1: gates = h @ W_hh^T + b_hh, then fused LSTM cell.
// grid: (B/16, H/16), block 32. Whh_t is [K=512][N=2048] bf16.
// ---------------------------------------------------------------------------
__global__ void __launch_bounds__(32)
k_gates_lstm(const bf16* __restrict__ h_bf, const bf16* __restrict__ Whh_t,
             const float* __restrict__ b_hh, float* __restrict__ c,
             bf16* __restrict__ ht_bf) {
  int lane = threadIdx.x;
  int mrow = lane & 15, half = lane >> 4;
  int m0 = blockIdx.x * 16;   // batch tile
  int n0 = blockIdx.y * 16;   // hidden tile
  v8f ai = {}, af = {}, ag = {}, ao = {};
  for (int k0 = 0; k0 < Hd; k0 += 32) {
    v16bf a = load_a_frag(h_bf + (m0 + mrow) * Hd + k0 + half * 8);
    const bf16* brow = Whh_t + (k0 + half * 8) * G4 + mrow;
    v16bf bi = load_b_frag(brow + n0, G4);
    v16bf bf_ = load_b_frag(brow + n0 + Hd, G4);
    v16bf bg = load_b_frag(brow + n0 + 2 * Hd, G4);
    v16bf bo = load_b_frag(brow + n0 + 3 * Hd, G4);
    ai = WMMA_BF16(a, bi, ai);
    af = WMMA_BF16(a, bf_, af);
    ag = WMMA_BF16(a, bg, ag);
    ao = WMMA_BF16(a, bo, ao);
  }
  int col = n0 + mrow;
  float bi_ = b_hh[col], bf2 = b_hh[col + Hd], bg_ = b_hh[col + 2 * Hd], bo_ = b_hh[col + 3 * Hd];
#pragma unroll
  for (int r = 0; r < 8; ++r) {
    int m = m0 + r + half * 8;
    float ig = ai[r] + bi_;
    float fg = af[r] + bf2;
    float gg = ag[r] + bg_;
    float og = ao[r] + bo_;
    float cold = c[m * Hd + col];
    float cnew = fsigm(fg) * cold + fsigm(ig) * ftanh(gg);
    float ht   = fsigm(og) * ftanh(cnew);
    c[m * Hd + col]    = cnew;          // c stays fp32 across all 128 steps
    ht_bf[m * Hd + col] = (bf16)ht;
  }
}

// ---------------------------------------------------------------------------
// K2: inp = h_t @ W_in^T + b_in.  grid (B/16, H/16), block 32.
// ---------------------------------------------------------------------------
__global__ void __launch_bounds__(32)
k_gemm_inp(const bf16* __restrict__ ht_bf, const bf16* __restrict__ Win_t,
           const float* __restrict__ b_in, float* __restrict__ inp) {
  int lane = threadIdx.x;
  int mrow = lane & 15, half = lane >> 4;
  int m0 = blockIdx.x * 16;
  int n0 = blockIdx.y * 16;
  v8f acc = {};
  for (int k0 = 0; k0 < Hd; k0 += 32) {
    v16bf a  = load_a_frag(ht_bf + (m0 + mrow) * Hd + k0 + half * 8);
    v16bf bb = load_b_frag(Win_t + (k0 + half * 8) * Hd + n0 + mrow, Hd);
    acc = WMMA_BF16(a, bb, acc);
  }
  int col = n0 + mrow;
#pragma unroll
  for (int r = 0; r < 8; ++r) {
    int m = m0 + r + half * 8;
    inp[m * Hd + col] = acc[r] + b_in[col];
  }
}

// ---------------------------------------------------------------------------
// K3: per-batch fused attention: scores (tanh-heavy), mask, softmax, write
// alpha row, argmax pointer, mask update, hid = ctx @ alpha.
// grid: B blocks of 128 threads (thread = position l).
// ---------------------------------------------------------------------------
__global__ void __launch_bounds__(128)
k_attn(const float* __restrict__ inp, const float* __restrict__ ctx,
       const float* __restrict__ V, float* __restrict__ mask,
       bf16* __restrict__ hid_bf, float* __restrict__ out_alpha,
       float* __restrict__ out_ptr, int step) {
  __shared__ float inp_s[Hd];
  __shared__ float V_s[Hd];
  __shared__ float red[Ld];
  __shared__ float alpha_s[Ld];
  __shared__ float s_max, s_sum;
  __shared__ int   s_idx;

  int tid = threadIdx.x;      // l position, 0..127
  int b   = blockIdx.x;
  for (int h = tid; h < Hd; h += Ld) { inp_s[h] = inp[b * Hd + h]; V_s[h] = V[h]; }
  __syncthreads();

  const float* ctxb = ctx + b * Hd * Ld;
  float acc = 0.0f;
#pragma unroll 4
  for (int h = 0; h < Hd; ++h)
    acc += V_s[h] * ftanh(inp_s[h] + ctxb[h * Ld + tid]);   // coalesced over l

  float mval = mask[b * Ld + tid];
  float sc = (mval == 0.0f) ? -3.4e38f : acc;

  red[tid] = sc; __syncthreads();
  if (tid == 0) {
    float m = red[0];
    for (int i = 1; i < Ld; ++i) m = fmaxf(m, red[i]);
    s_max = m;
  }
  __syncthreads();
  float e = fexp2((sc - s_max) * LOG2E);
  red[tid] = e; __syncthreads();
  if (tid == 0) {
    float s = 0.0f;
    for (int i = 0; i < Ld; ++i) s += red[i];
    s_sum = s;
  }
  __syncthreads();
  float alpha = e * frcp(s_sum);
  alpha_s[tid] = alpha;
  out_alpha[b * Ld * Ld + step * Ld + tid] = alpha;

  red[tid] = alpha * mval; __syncthreads();
  if (tid == 0) {
    float m = red[0]; int bi = 0;
    for (int i = 1; i < Ld; ++i) if (red[i] > m) { m = red[i]; bi = i; }  // first-max tie-break
    s_idx = bi;
    out_ptr[b * Ld + step] = (float)bi;
  }
  __syncthreads();
  if (tid == s_idx) mask[b * Ld + tid] = 0.0f;   // mask *= (1 - onehot(idx))
  __syncthreads();

  // hid[b,h] = sum_l ctx[b,h,l] * alpha[l]  (fp32 accumulate, store bf16 for GEMM)
  for (int h = tid; h < Hd; h += Ld) {
    const float* row = ctxb + h * Ld;
    float a2 = 0.0f;
#pragma unroll 4
    for (int l = 0; l < Ld; ++l) a2 += row[l] * alpha_s[l];
    hid_bf[b * Hd + h] = (bf16)a2;
  }
}

// ---------------------------------------------------------------------------
// K4: h_new = tanh(concat[hid, h_t] @ W_out^T + b_out).
// grid (B/16, H/16), block 32. Wout_t is [K=1024][N=512] bf16.
// ---------------------------------------------------------------------------
__global__ void __launch_bounds__(32)
k_hout(const bf16* __restrict__ hid_bf, const bf16* __restrict__ ht_bf,
       const bf16* __restrict__ Wout_t, const float* __restrict__ b_out,
       float* __restrict__ h, bf16* __restrict__ h_bf) {
  int lane = threadIdx.x;
  int mrow = lane & 15, half = lane >> 4;
  int m0 = blockIdx.x * 16;
  int n0 = blockIdx.y * 16;
  v8f acc = {};
  for (int k0 = 0; k0 < K2H; k0 += 32) {
    const bf16* asrc = (k0 < Hd) ? (hid_bf + (m0 + mrow) * Hd + k0)
                                 : (ht_bf  + (m0 + mrow) * Hd + (k0 - Hd));
    v16bf a  = load_a_frag(asrc + half * 8);
    v16bf bb = load_b_frag(Wout_t + (k0 + half * 8) * Hd + n0 + mrow, Hd);
    acc = WMMA_BF16(a, bb, acc);
  }
  int col = n0 + mrow;
#pragma unroll
  for (int r = 0; r < 8; ++r) {
    int m = m0 + r + half * 8;
    float v = ftanh(acc[r] + b_out[col]);
    h[m * Hd + col]    = v;
    h_bf[m * Hd + col] = (bf16)v;
  }
}

// ---------------------------------------------------------------------------
// Host launcher
// ---------------------------------------------------------------------------
extern "C" void kernel_launch(void* const* d_in, const int* in_sizes, int n_in,
                              void* d_out, int out_size, void* d_ws, size_t ws_size,
                              hipStream_t stream) {
  (void)in_sizes; (void)n_in; (void)out_size; (void)ws_size;
  const float* h0      = (const float*)d_in[2];
  const float* c0      = (const float*)d_in[3];
  const float* context = (const float*)d_in[4];
  const float* W_hh    = (const float*)d_in[5];
  const float* b_hh    = (const float*)d_in[6];
  const float* W_in    = (const float*)d_in[7];
  const float* b_in    = (const float*)d_in[8];
  const float* W_ctx   = (const float*)d_in[9];
  const float* b_ctx   = (const float*)d_in[10];
  const float* V       = (const float*)d_in[11];
  const float* W_out   = (const float*)d_in[12];
  const float* b_out   = (const float*)d_in[13];

  float* out       = (float*)d_out;
  float* out_alpha = out;                              // [B,L,L]
  float* out_ptr   = out_alpha + Bsz * Ld * Ld;        // [B,L]
  float* out_h     = out_ptr + Bsz * Ld;               // [B,H]
  float* out_c     = out_h + Bsz * Hd;                 // [B,H]

  char* ws = (char*)d_ws;
  size_t off = 0;
  auto walloc = [&](size_t bytes) -> void* {
    void* p = ws + off;
    off += (bytes + 255) & ~(size_t)255;
    return p;
  };
  float* ctx      = (float*)walloc((size_t)Bsz * Hd * Ld * 4);   // 16 MB (L2-resident)
  bf16*  ctxin_bf = (bf16*) walloc((size_t)Bsz * Hd * Ld * 2);   // 8 MB
  bf16*  Whh_t    = (bf16*) walloc((size_t)Hd * G4 * 2);         // 2 MB
  bf16*  Win_t    = (bf16*) walloc((size_t)Hd * Hd * 2);
  bf16*  Wout_t   = (bf16*) walloc((size_t)K2H * Hd * 2);
  bf16*  Wctx_bf  = (bf16*) walloc((size_t)Hd * Hd * 2);
  float* hbuf     = (float*)walloc((size_t)Bsz * Hd * 4);
  bf16*  h_bf     = (bf16*) walloc((size_t)Bsz * Hd * 2);
  float* cbuf     = (float*)walloc((size_t)Bsz * Hd * 4);
  bf16*  ht_bf    = (bf16*) walloc((size_t)Bsz * Hd * 2);
  bf16*  hid_bf   = (bf16*) walloc((size_t)Bsz * Hd * 2);
  float* inp      = (float*)walloc((size_t)Bsz * Hd * 4);
  float* mask     = (float*)walloc((size_t)Bsz * Ld * 4);

  // --- prep: bf16 weight transposes ([K][N] layout => contiguous B-frag rows)
  k_transpose_bf<<<(G4 * Hd + 255) / 256, 256, 0, stream>>>(W_hh, Whh_t, G4, Hd);
  k_transpose_bf<<<(Hd * Hd + 255) / 256, 256, 0, stream>>>(W_in, Win_t, Hd, Hd);
  k_transpose_bf<<<(Hd * K2H + 255) / 256, 256, 0, stream>>>(W_out, Wout_t, Hd, K2H);
  k_cvt_bf<<<(Hd * Hd + 255) / 256, 256, 0, stream>>>(W_ctx, Wctx_bf, Hd * Hd);
  k_cvt_bf<<<(Bsz * Hd * Ld + 255) / 256, 256, 0, stream>>>(context, ctxin_bf, Bsz * Hd * Ld);
  k_init_state<<<(Bsz * Hd + 255) / 256, 256, 0, stream>>>(h0, c0, hbuf, cbuf, h_bf, mask);

  // --- step-invariant ctx projection (WMMA)
  dim3 gctx(Ld / 16, Hd / 16, Bsz);
  k_ctx_gemm<<<gctx, 32, 0, stream>>>(Wctx_bf, ctxin_bf, b_ctx, ctx);

  // --- 128 sequential decode steps
  dim3 gtile(Bsz / 16, Hd / 16);
  for (int step = 0; step < Ld; ++step) {
    k_gates_lstm<<<gtile, 32, 0, stream>>>(h_bf, Whh_t, b_hh, cbuf, ht_bf);
    k_gemm_inp<<<gtile, 32, 0, stream>>>(ht_bf, Win_t, b_in, inp);
    k_attn<<<Bsz, Ld, 0, stream>>>(inp, ctx, V, mask, hid_bf, out_alpha, out_ptr, step);
    k_hout<<<gtile, 32, 0, stream>>>(hid_bf, ht_bf, Wout_t, b_out, hbuf, h_bf);
  }

  k_finalize<<<(Bsz * Hd + 255) / 256, 256, 0, stream>>>(hbuf, cbuf, out_h, out_c);
}
